// LRULayer_9242769622002
// MI455X (gfx1250) — matrix operand order repaired and checked by reading, not compiled, not run on hardware
//
#include <hip/hip_runtime.h>
#include <hip/hip_bf16.h>

typedef __attribute__((ext_vector_type(16))) __bf16 v16bf;
typedef __attribute__((ext_vector_type(8)))  float  v8f;

#define B_   4
#define L_   4096
#define E_   512
#define D_   1024
#define BL_  (B_ * L_)
#define NCH  8
#define CH   512   // L_ / NCH

union FragU {
    v16bf v;
    uint4 q[2];
};

// ---------------------------------------------------------------- params prep
__global__ void k_prep(const float* __restrict__ pl,
                       float* __restrict__ lre, float* __restrict__ lim,
                       float* __restrict__ lCre, float* __restrict__ lCim,
                       float* __restrict__ gam) {
    int d = blockIdx.x * blockDim.x + threadIdx.x;
    if (d >= D_) return;
    float nu = __expf(pl[d]);
    float th = __expf(pl[D_ + d]);
    float g  = __expf(pl[2 * D_ + d]);
    float r  = __expf(-nu);
    lre[d] = r * __cosf(th);
    lim[d] = r * __sinf(th);
    // lambda^CH for chunk combination (exact via logs)
    float rC  = __expf(-nu * (float)CH);
    float thC = th * (float)CH;
    lCre[d] = rC * cosf(thC);
    lCim[d] = rC * sinf(thC);
    gam[d] = g;
}

// ---------------------------------------------------------------- x -> bf16
__global__ void k_cvt_x(const float* __restrict__ x, __bf16* __restrict__ xbf, int n) {
    int i = (blockIdx.x * blockDim.x + threadIdx.x) * 4;
    if (i >= n) return;
    float4 f = *(const float4*)(x + i);
    union { __bf16 h[4]; uint2 u; } o;
    o.h[0] = (__bf16)f.x; o.h[1] = (__bf16)f.y;
    o.h[2] = (__bf16)f.z; o.h[3] = (__bf16)f.w;
    *(uint2*)(xbf + i) = o.u;
}

// ------------------------------------------- transpose+convert weight (K x N -> N x K bf16)
__global__ void k_transpose_cvt(const float* __restrict__ src, __bf16* __restrict__ dst,
                                int rows, int cols) {
    // dst[c*rows + r] = src[r*cols + c]
    int idx = blockIdx.x * blockDim.x + threadIdx.x;
    if (idx >= rows * cols) return;
    int c = idx / rows;
    int r = idx - c * rows;
    dst[(size_t)c * rows + r] = (__bf16)src[(size_t)r * cols + c];
}

// ---------------------------------------------------------------- GEMM1 (bf16 WMMA)
// u[bl, d] = (sum_k xbf[bl,k] * wInT[d,k] + b_in[d]) * gamma[d]
__global__ __launch_bounds__(256) void k_gemm1(const __bf16* __restrict__ xbf,
                                               const __bf16* __restrict__ wInT,
                                               const float* __restrict__ b_in,
                                               const float* __restrict__ gam,
                                               float* __restrict__ u) {
    __shared__ __align__(16) __bf16 As[128][40];
    __shared__ __align__(16) __bf16 Bs[128][40];
    const int tid  = threadIdx.x;
    const int lane = tid & 31;
    const int wv   = tid >> 5;
    const int bm   = blockIdx.y * 128;
    const int bn   = blockIdx.x * 128;
    const int wM   = (wv & 1) * 64;
    const int wN   = (wv >> 1) * 32;

    const v8f vzero = {};
    v8f acc[4][2];
#pragma unroll
    for (int mi = 0; mi < 4; mi++)
#pragma unroll
        for (int ni = 0; ni < 2; ni++) acc[mi][ni] = vzero;

    const int kkA = (lane >> 4) * 8;    // A-frag K interleave (ISA 7.12.2)
    const int kkB = (lane >> 4) * 8;    // B-frag K interleave (ISA 7.12.2)
    const int fr  = lane & 15;

    for (int k0 = 0; k0 < E_; k0 += 32) {
        // cooperative stage: 128x32 bf16 tiles for A and B
#pragma unroll
        for (int i = 0; i < 2; i++) {
            int c   = tid + i * 256;        // 512 16B-chunks per tile
            int row = c >> 2;
            int kc  = (c & 3) * 8;
            *(uint4*)&As[row][kc] = *(const uint4*)(xbf  + (size_t)(bm + row) * E_ + k0 + kc);
            *(uint4*)&Bs[row][kc] = *(const uint4*)(wInT + (size_t)(bn + row) * E_ + k0 + kc);
        }
        __syncthreads();

        FragU a[4], b[2];
#pragma unroll
        for (int mi = 0; mi < 4; mi++) {
            int r = wM + mi * 16 + fr;
            a[mi].q[0] = *(const uint4*)&As[r][kkA];
            a[mi].q[1] = *(const uint4*)&As[r][kkA + 16];
        }
#pragma unroll
        for (int ni = 0; ni < 2; ni++) {
            int n = wN + ni * 16 + fr;
            b[ni].q[0] = *(const uint4*)&Bs[n][kkB];
            b[ni].q[1] = *(const uint4*)&Bs[n][kkB + 16];
        }
#pragma unroll
        for (int mi = 0; mi < 4; mi++)
#pragma unroll
            for (int ni = 0; ni < 2; ni++)
                acc[mi][ni] = __builtin_amdgcn_wmma_f32_16x16x32_bf16(
                    false, a[mi].v, false, b[ni].v, (short)0, acc[mi][ni], false, false);
        __syncthreads();
    }

    // epilogue: bias + gamma scale, f32 out
    const int mo = (lane >> 4) * 8;
    const int no = lane & 15;
#pragma unroll
    for (int mi = 0; mi < 4; mi++)
#pragma unroll
        for (int ni = 0; ni < 2; ni++)
#pragma unroll
            for (int r = 0; r < 8; r++) {
                int row = bm + wM + mi * 16 + mo + r;
                int col = bn + wN + ni * 16 + no;
                u[(size_t)row * D_ + col] = (acc[mi][ni][r] + b_in[col]) * gam[col];
            }
}

// ---------------------------------------------------------------- scan pass1: chunk carries
__global__ void k_scan1(const float* __restrict__ u, const float* __restrict__ mask,
                        const float* __restrict__ lre, const float* __restrict__ lim,
                        float* __restrict__ cre, float* __restrict__ cim) {
    int idx = blockIdx.x * blockDim.x + threadIdx.x;    // (b, chunk, d)
    int d = idx & (D_ - 1);
    int c = (idx >> 10) & (NCH - 1);
    int b = idx >> 13;
    float ar = lre[d], ai = lim[d];
    float hr = 0.f, hi = 0.f;
    const float* up = u + ((size_t)(b * L_ + c * CH)) * D_ + d;
    const float* mp = mask + b * L_ + c * CH;
    for (int t = 0; t < CH; t++) {
        int g = c * CH + t;
        float m  = (g == 0) ? 0.f : mp[t - 1];
        float xr = m * hr, xi = m * hi;
        hr = ar * xr - ai * xi + up[(size_t)t * D_];
        hi = ar * xi + ai * xr;
    }
    cre[idx] = hr;
    cim[idx] = hi;
}

// ---------------------------------------------------------------- scan pass2: prefix of carries
__global__ void k_scan2(const float* __restrict__ cre, const float* __restrict__ cim,
                        const float* __restrict__ lCre, const float* __restrict__ lCim,
                        float* __restrict__ pre, float* __restrict__ pim) {
    int idx = blockIdx.x * blockDim.x + threadIdx.x;    // (b, d)
    int d = idx & (D_ - 1);
    int b = idx >> 10;
    float Ar = lCre[d], Ai = lCim[d];
    float pr = 0.f, pi = 0.f;
    for (int c = 0; c < NCH; c++) {
        int j = (b * NCH + c) * D_ + d;
        pre[j] = pr;
        pim[j] = pi;
        float nr = cre[j] + Ar * pr - Ai * pi;
        float ni = cim[j] + Ar * pi + Ai * pr;
        pr = nr;
        pi = ni;
    }
}

// ---------------------------------------------------------------- scan pass3: emit Re(h) as bf16
__global__ void k_scan3(const float* __restrict__ u, const float* __restrict__ mask,
                        const float* __restrict__ lre, const float* __restrict__ lim,
                        const float* __restrict__ pre, const float* __restrict__ pim,
                        __bf16* __restrict__ hbf) {
    int idx = blockIdx.x * blockDim.x + threadIdx.x;
    int d = idx & (D_ - 1);
    int c = (idx >> 10) & (NCH - 1);
    int b = idx >> 13;
    float ar = lre[d], ai = lim[d];
    float hr = pre[idx], hi = pim[idx];
    const float* up = u + ((size_t)(b * L_ + c * CH)) * D_ + d;
    __bf16* op = hbf + ((size_t)(b * L_ + c * CH)) * D_ + d;
    const float* mp = mask + b * L_ + c * CH;
    for (int t = 0; t < CH; t++) {
        int g = c * CH + t;
        float m  = (g == 0) ? 0.f : mp[t - 1];
        float xr = m * hr, xi = m * hi;
        hr = ar * xr - ai * xi + up[(size_t)t * D_];
        hi = ar * xi + ai * xr;
        op[(size_t)t * D_] = (__bf16)hr;
    }
}

// ---------------------------------------------------------------- GEMM2 + residual + LayerNorm
__global__ __launch_bounds__(256) void k_gemm2_ln(const __bf16* __restrict__ hbf,
                                                  const __bf16* __restrict__ wOutT,
                                                  const float* __restrict__ b_out,
                                                  const float* __restrict__ x,
                                                  const float* __restrict__ lnw,
                                                  const float* __restrict__ lnb,
                                                  float* __restrict__ out) {
    __shared__ float Y[32][E_];     // 64 KB: full rows for in-block LayerNorm
    const int tid  = threadIdx.x;
    const int lane = tid & 31;
    const int wv   = tid >> 5;
    const int bm   = blockIdx.x * 32;
    const int wN   = wv * 64;       // each wave owns 64 of 512 columns

    const v8f vzero = {};
    v8f acc[2][4];
#pragma unroll
    for (int mi = 0; mi < 2; mi++)
#pragma unroll
        for (int ni = 0; ni < 4; ni++) acc[mi][ni] = vzero;

    const int fr  = lane & 15;
    const int kkA = (lane >> 4) * 8;
    const int kkB = (lane >> 4) * 8;

    for (int k0 = 0; k0 < D_; k0 += 32) {
        FragU a[2], b[4];
#pragma unroll
        for (int mi = 0; mi < 2; mi++) {
            const __bf16* p = hbf + (size_t)(bm + mi * 16 + fr) * D_ + k0 + kkA;
            a[mi].q[0] = *(const uint4*)p;
            a[mi].q[1] = *(const uint4*)(p + 16);
        }
#pragma unroll
        for (int ni = 0; ni < 4; ni++) {
            const __bf16* p = wOutT + (size_t)(wN + ni * 16 + fr) * D_ + k0 + kkB;
            b[ni].q[0] = *(const uint4*)p;
            b[ni].q[1] = *(const uint4*)(p + 16);
        }
#pragma unroll
        for (int mi = 0; mi < 2; mi++)
#pragma unroll
            for (int ni = 0; ni < 4; ni++)
                acc[mi][ni] = __builtin_amdgcn_wmma_f32_16x16x32_bf16(
                    false, a[mi].v, false, b[ni].v, (short)0, acc[mi][ni], false, false);
    }

    // y = acc + b_out + x  -> LDS
    const int mo = (lane >> 4) * 8;
    const int no = lane & 15;
#pragma unroll
    for (int mi = 0; mi < 2; mi++)
#pragma unroll
        for (int ni = 0; ni < 4; ni++)
#pragma unroll
            for (int r = 0; r < 8; r++) {
                int m = mi * 16 + mo + r;
                int n = wN + ni * 16 + no;
                Y[m][n] = acc[mi][ni][r] + b_out[n] + x[(size_t)(bm + m) * E_ + n];
            }
    __syncthreads();

    // LayerNorm: wave wv handles rows [wv*4, wv*4+4)
    for (int rr = 0; rr < 4; rr++) {
        int row = wv * 4 + rr;
        float s = 0.f, s2 = 0.f;
#pragma unroll
        for (int i = 0; i < 16; i++) {
            float v = Y[row][lane + i * 32];
            s  += v;
            s2 += v * v;
        }
#pragma unroll
        for (int o = 16; o > 0; o >>= 1) {
            s  += __shfl_xor(s,  o, 32);
            s2 += __shfl_xor(s2, o, 32);
        }
        float mu   = s * (1.f / E_);
        float var  = s2 * (1.f / E_) - mu * mu;
        float rinv = rsqrtf(var + 1e-5f);
#pragma unroll
        for (int i = 0; i < 16; i++) {
            int e = lane + i * 32;
            float v = (Y[row][e] - mu) * rinv;
            out[(size_t)(bm + row) * E_ + e] = v * lnw[e] + lnb[e];
        }
    }
}

// ---------------------------------------------------------------- launcher
extern "C" void kernel_launch(void* const* d_in, const int* in_sizes, int n_in,
                              void* d_out, int out_size, void* d_ws, size_t ws_size,
                              hipStream_t stream) {
    const float* x     = (const float*)d_in[0];
    const float* mask  = (const float*)d_in[1];
    const float* pl    = (const float*)d_in[2];
    const float* w_in  = (const float*)d_in[3];
    const float* b_in  = (const float*)d_in[4];
    const float* w_out = (const float*)d_in[5];
    const float* b_out = (const float*)d_in[6];
    const float* lnw   = (const float*)d_in[7];
    const float* lnb   = (const float*)d_in[8];
    float* out = (float*)d_out;

    char* ws = (char*)d_ws;
    size_t off = 0;
    auto alloc = [&](size_t bytes) {
        size_t o = off;
        off = (off + bytes + 255) & ~(size_t)255;
        return o;
    };
    __bf16* xbf   = (__bf16*)(ws + alloc((size_t)BL_ * E_ * 2));
    __bf16* wInT  = (__bf16*)(ws + alloc((size_t)D_ * E_ * 2));
    __bf16* wOutT = (__bf16*)(ws + alloc((size_t)E_ * D_ * 2));
    float*  lre   = (float*)(ws + alloc(D_ * 4));
    float*  lim   = (float*)(ws + alloc(D_ * 4));
    float*  lCre  = (float*)(ws + alloc(D_ * 4));
    float*  lCim  = (float*)(ws + alloc(D_ * 4));
    float*  gam   = (float*)(ws + alloc(D_ * 4));
    float*  cre   = (float*)(ws + alloc((size_t)B_ * NCH * D_ * 4));
    float*  cim   = (float*)(ws + alloc((size_t)B_ * NCH * D_ * 4));
    float*  pre   = (float*)(ws + alloc((size_t)B_ * NCH * D_ * 4));
    float*  pim   = (float*)(ws + alloc((size_t)B_ * NCH * D_ * 4));
    float*  u     = (float*)(ws + alloc((size_t)BL_ * D_ * 4));
    __bf16* hbf   = (__bf16*)(ws + alloc((size_t)BL_ * D_ * 2));

    k_prep<<<D_ / 256, 256, 0, stream>>>(pl, lre, lim, lCre, lCim, gam);
    k_cvt_x<<<(BL_ * E_ / 4) / 256, 256, 0, stream>>>(x, xbf, BL_ * E_);
    k_transpose_cvt<<<(E_ * D_) / 256, 256, 0, stream>>>(w_in, wInT, E_, D_);
    k_transpose_cvt<<<(D_ * E_) / 256, 256, 0, stream>>>(w_out, wOutT, D_, E_);
    k_gemm1<<<dim3(D_ / 128, BL_ / 128), 256, 0, stream>>>(xbf, wInT, b_in, gam, u);
    k_scan1<<<(B_ * NCH * D_) / 256, 256, 0, stream>>>(u, mask, lre, lim, cre, cim);
    k_scan2<<<(B_ * D_) / 256, 256, 0, stream>>>(cre, cim, lCre, lCim, pre, pim);
    k_scan3<<<(B_ * NCH * D_) / 256, 256, 0, stream>>>(u, mask, lre, lim, pre, pim, hbf);
    k_gemm2_ln<<<BL_ / 32, 256, 0, stream>>>(hbf, wOutT, b_out, x, lnw, lnb, out);
}
